// hidden_lstm_6734508720798
// MI455X (gfx1250) — compile-verified, matrix operand order checked by reading, not run on hardware
//
#include <hip/hip_runtime.h>

// LSTM: B=64, T=512, I=H=1024, L=3.
// Phase 1 per layer: bf16 WMMA GEMM  x_proj = X @ W_ih^T + b_ih + b_hh.
// Phase 2 per layer: ONE persistent kernel runs all 512 timesteps with a
// grid-wide barrier; W_hh slices live in LDS for the whole sequence (loaded
// via the gfx1250 async-to-LDS path when available), cell state c lives in
// registers, h is exchanged through global/L2.

typedef __attribute__((ext_vector_type(16))) __bf16 v16bf;
typedef __attribute__((ext_vector_type(8)))  __bf16 v8bf;
typedef __attribute__((ext_vector_type(8)))  float  v8f;
typedef __attribute__((ext_vector_type(4)))  int    v4i;

#define BB 64
#define TT 512
#define HH 1024
#define G4 4096          // 4*H
#define NSLICE 16        // h-columns per block in the scan
#define WPAD (HH + 8)    // padded LDS row stride (spreads DS banks)

#if defined(__AMDGCN__) && __has_builtin(__builtin_amdgcn_global_load_async_to_lds_b128)
#define HAVE_ASYNC_LDS 1
#else
#define HAVE_ASYNC_LDS 0
#endif

__device__ __forceinline__ void wait_asynccnt0() {
#if defined(__AMDGCN__)
#if __has_builtin(__builtin_amdgcn_s_wait_asynccnt)
  __builtin_amdgcn_s_wait_asynccnt(0);
#elif HAVE_ASYNC_LDS
  asm volatile("s_wait_asynccnt 0" ::: "memory");
#endif
#endif
}

// ---------- bf16 helpers --------------------------------------------------------
__device__ __forceinline__ __bf16 f2bf(float f) {
  unsigned u = __builtin_bit_cast(unsigned, f);
  unsigned r = (u + 0x7FFFu + ((u >> 16) & 1u)) >> 16;
  unsigned short s = (unsigned short)r;
  return __builtin_bit_cast(__bf16, s);
}

// ---------- 16x32 bf16 fragment load (global or LDS, row-major, stride elems) ---
// ISA layout (CDNA5 ISA §7.12.2): lane holds row (lane&15); lanes 0-15 take
// K-groups {0..7,16..23}, lanes 16-31 take {8..15,24..31}.
__device__ __forceinline__ v16bf load_frag16x32(const __bf16* base, int stride) {
  int lane = threadIdx.x & 31;
  int r    = lane & 15;
  int ks   = (lane >> 4) << 3;            // 0 or 8
  const __bf16* p = base + (size_t)r * stride + ks;
  v8bf lo = *(const v8bf*)(p);
  v8bf hi = *(const v8bf*)(p + 16);
  v16bf f;
#pragma unroll
  for (int i = 0; i < 8; ++i) { f[i] = lo[i]; f[i + 8] = hi[i]; }
  return f;
}

// ---------- small utility kernels -----------------------------------------------
__global__ void cvt_f32_bf16(const float* __restrict__ in,
                             __bf16* __restrict__ out, int n) {
  int i = blockIdx.x * blockDim.x + threadIdx.x;
  if (i < n) out[i] = f2bf(in[i]);
}

__global__ void reset_bar(int* bar) { *bar = 0; }

// ---------- input projection: xproj = Xbf @ W_ih^T + b_ih + b_hh ----------------
// A: [32768,1024] bf16  W: [4096,1024] bf16  out: [32768,4096] fp32
__global__ __launch_bounds__(256)
void lstm_xproj_gemm(const __bf16* __restrict__ A,
                     const __bf16* __restrict__ W,
                     const float* __restrict__ bih,
                     const float* __restrict__ bhh,
                     float* __restrict__ out) {
  const int K = HH;
  int wave = threadIdx.x >> 5;
  int lane = threadIdx.x & 31;
  int mw = wave & 3;
  int nw = wave >> 2;
  int m0 = blockIdx.x * 128 + mw * 32;
  int n0 = blockIdx.y * 128 + nw * 64;

  v8f acc[2][4];
  v8f zero = {};
#pragma unroll
  for (int i = 0; i < 2; ++i)
#pragma unroll
    for (int j = 0; j < 4; ++j) acc[i][j] = zero;

  for (int kk = 0; kk < K; kk += 32) {
    v16bf a[2], b[4];
#pragma unroll
    for (int i = 0; i < 2; ++i)
      a[i] = load_frag16x32(A + (size_t)(m0 + i * 16) * K + kk, K);
#pragma unroll
    for (int j = 0; j < 4; ++j)
      b[j] = load_frag16x32(W + (size_t)(n0 + j * 16) * K + kk, K);
#pragma unroll
    for (int i = 0; i < 2; ++i)
#pragma unroll
      for (int j = 0; j < 4; ++j)
        acc[i][j] = __builtin_amdgcn_wmma_f32_16x16x32_bf16(
            false, a[i], false, b[j], (short)0, acc[i][j], false, false);
  }

  int ln = lane & 15;
  int mh = (lane >> 4) << 3;
#pragma unroll
  for (int i = 0; i < 2; ++i) {
#pragma unroll
    for (int j = 0; j < 4; ++j) {
      int n = n0 + j * 16 + ln;
      float bias = bih[n] + bhh[n];
#pragma unroll
      for (int r = 0; r < 8; ++r) {
        int m = m0 + i * 16 + mh + r;
        out[(size_t)m * G4 + n] = acc[i][j][r] + bias;
      }
    }
  }
}

// ---------- persistent recurrent scan -------------------------------------------
// grid = 64 blocks (16 h-columns each) x 128 threads (4 waves).
// Wave w owns M-tile w (batches w*16..w*16+15) for ALL 4 gates:
//   A frag (h) loaded once per wave from global, B frags from LDS-resident W_hh.
// Cell state c lives in registers (8 elems/thread) for the whole sequence.
__global__ __launch_bounds__(128)
void lstm_scan(const __bf16* __restrict__ Whh,     // [4096,1024] bf16 (global)
               const float* __restrict__ xproj,    // [B*T,4096] fp32
               __bf16* __restrict__ h0,            // [64,1024] bf16 (t even in)
               __bf16* __restrict__ h1,            // [64,1024] bf16 (t odd  in)
               float* __restrict__ out_f32,        // [B,T,H] or null
               __bf16* __restrict__ out_bf,        // [B,T,H] bf16 or null
               int* __restrict__ bar) {
  __shared__ __bf16 wlds[4][NSLICE][WPAD];   // 132 KB: this block's W_hh slices
  __shared__ float  gbuf[4][BB][NSLICE];     // 16 KB: gate exchange

  const int tid  = threadIdx.x;
  const int wave = tid >> 5;
  const int lane = tid & 31;
  const int n0   = blockIdx.x * NSLICE;      // h-column slice base
  const int nblk = gridDim.x;

  // ---- preload W_hh gate slices into LDS (once) --------------------------------
  // chunks of 8 bf16 (16 B); total 4*16*(1024/8) = 8192 chunks, 64 per thread.
  // Uses GLOBAL_LOAD_ASYNC_TO_LDS_B128 (ASYNCcnt-tracked) when the toolchain
  // exposes it; otherwise a plain register-bounce copy.
  for (int c = tid; c < 4 * NSLICE * (HH / 8); c += 128) {
    int g  = c / (NSLICE * (HH / 8));
    int rr = (c / (HH / 8)) % NSLICE;
    int kb = c % (HH / 8);
    const __bf16* src = Whh + (size_t)(g * HH + n0 + rr) * HH + kb * 8;
    __bf16* dst = &wlds[g][rr][kb * 8];
#if HAVE_ASYNC_LDS
    __builtin_amdgcn_global_load_async_to_lds_b128(
        (__attribute__((address_space(1))) v4i*)src,
        (__attribute__((address_space(3))) v4i*)dst, 0, 0);
#else
    *(v8bf*)dst = *(const v8bf*)src;
#endif
  }
  wait_asynccnt0();   // this wave's async copies into LDS are complete

  // ---- zero this block's slice of h0, init c in registers ----------------------
  float creg[8];
#pragma unroll
  for (int j = 0; j < 8; ++j) {
    creg[j] = 0.0f;
    int idx = tid + j * 128;                 // 0..1023
    int b = idx >> 4, nl = idx & 15;
    h0[b * HH + n0 + nl] = f2bf(0.0f);
  }

  // ---- grid barrier (release/acquire on L2 atomic) -----------------------------
  int barTarget = 0;
  auto grid_sync = [&](void) {
    __syncthreads();
    barTarget += nblk;
    if (tid == 0) {
      __hip_atomic_fetch_add(bar, 1, __ATOMIC_RELEASE, __HIP_MEMORY_SCOPE_AGENT);
      while (__hip_atomic_load(bar, __ATOMIC_ACQUIRE, __HIP_MEMORY_SCOPE_AGENT)
             < barTarget) {
        __builtin_amdgcn_s_sleep(2);
      }
    }
    __syncthreads();
  };
  grid_sync();   // h0 + LDS weights ready everywhere

  for (int t = 0; t < TT; ++t) {
    const __bf16* hp = (t & 1) ? h1 : h0;
    __bf16*       hn = (t & 1) ? h0 : h1;

    // ---- gates = h_prev @ W_hh^T for this wave's 16 batches, all 4 gates ------
    v8f acc[4];
    v8f zero = {};
#pragma unroll
    for (int g = 0; g < 4; ++g) acc[g] = zero;

    for (int kk = 0; kk < HH; kk += 32) {
      v16bf afrag = load_frag16x32(hp + (size_t)(wave * 16) * HH + kk, HH);
#pragma unroll
      for (int g = 0; g < 4; ++g) {
        v16bf bfrag = load_frag16x32(&wlds[g][0][0] + kk, WPAD);
        acc[g] = __builtin_amdgcn_wmma_f32_16x16x32_bf16(
            false, afrag, false, bfrag, (short)0, acc[g], false, false);
      }
    }

    // ---- exchange gates through LDS -------------------------------------------
    const int ln = lane & 15;
    const int mh = (lane >> 4) << 3;
#pragma unroll
    for (int g = 0; g < 4; ++g)
#pragma unroll
      for (int r = 0; r < 8; ++r)
        gbuf[g][wave * 16 + mh + r][ln] = acc[g][r];
    __syncthreads();

    // ---- LSTM cell update (c in registers) ------------------------------------
#pragma unroll
    for (int j = 0; j < 8; ++j) {
      int idx = tid + j * 128;
      int b = idx >> 4, nl = idx & 15;
      int n = n0 + nl;
      size_t xrow = (size_t)(b * TT + t) * G4;
      float ig = gbuf[0][b][nl] + xproj[xrow + 0 * HH + n];
      float fg = gbuf[1][b][nl] + xproj[xrow + 1 * HH + n];
      float gg = gbuf[2][b][nl] + xproj[xrow + 2 * HH + n];
      float og = gbuf[3][b][nl] + xproj[xrow + 3 * HH + n];
      if (t + 1 < TT) {  // prefetch next timestep's x_proj row (global_prefetch_b8)
        __builtin_prefetch(&xproj[xrow + G4 + n], 0, 1);
        __builtin_prefetch(&xproj[xrow + G4 + HH + n], 0, 1);
        __builtin_prefetch(&xproj[xrow + G4 + 2 * HH + n], 0, 1);
        __builtin_prefetch(&xproj[xrow + G4 + 3 * HH + n], 0, 1);
      }
      ig = 1.0f / (1.0f + __expf(-ig));
      fg = 1.0f / (1.0f + __expf(-fg));
      og = 1.0f / (1.0f + __expf(-og));
      gg = tanhf(gg);
      float cn = fg * creg[j] + ig * gg;
      creg[j] = cn;
      float hv = og * tanhf(cn);
      hn[b * HH + n] = f2bf(hv);
      size_t orow = (size_t)(b * TT + t) * HH + n;
      if (out_f32) out_f32[orow] = hv;
      if (out_bf)  out_bf[orow]  = f2bf(hv);
    }

    grid_sync();   // h_next visible device-wide; gbuf safe to overwrite
  }
}

// ---------- host orchestration --------------------------------------------------
extern "C" void kernel_launch(void* const* d_in, const int* in_sizes, int n_in,
                              void* d_out, int out_size, void* d_ws, size_t ws_size,
                              hipStream_t stream) {
  const float* x   = (const float*)d_in[0];   // [B,T,I]
  const float* Wih = (const float*)d_in[1];   // [L,4H,I]
  const float* Whh = (const float*)d_in[2];   // [L,4H,H]
  const float* bih = (const float*)d_in[3];   // [L,4H]
  const float* bhh = (const float*)d_in[4];   // [L,4H]
  float* out = (float*)d_out;                 // [B,T,H]

  char* ws = (char*)d_ws;
  const size_t actElems = (size_t)BB * TT * HH;
  const size_t wElems   = (size_t)G4 * HH;
  __bf16* act0  = (__bf16*)ws;  ws += actElems * 2;
  __bf16* act1  = (__bf16*)ws;  ws += actElems * 2;
  __bf16* wihb  = (__bf16*)ws;  ws += wElems * 2;
  __bf16* whhb  = (__bf16*)ws;  ws += wElems * 2;
  float*  xproj = (float*)ws;   ws += (size_t)BB * TT * G4 * 4;
  __bf16* h0    = (__bf16*)ws;  ws += (size_t)BB * HH * 2;
  __bf16* h1    = (__bf16*)ws;  ws += (size_t)BB * HH * 2;
  int*    bar   = (int*)ws;     ws += 256;

  {
    int n = (int)actElems;
    cvt_f32_bf16<<<(n + 255) / 256, 256, 0, stream>>>(x, act0, n);
  }

  for (int l = 0; l < 3; ++l) {
    const __bf16* actin = (l & 1) ? act1 : act0;
    __bf16* actout      = (l & 1) ? act0 : act1;
    const bool last = (l == 2);

    int wn = (int)wElems;
    cvt_f32_bf16<<<(wn + 255) / 256, 256, 0, stream>>>(Wih + (size_t)l * wElems, wihb, wn);
    cvt_f32_bf16<<<(wn + 255) / 256, 256, 0, stream>>>(Whh + (size_t)l * wElems, whhb, wn);

    lstm_xproj_gemm<<<dim3((BB * TT) / 128, G4 / 128), 256, 0, stream>>>(
        actin, wihb, bih + (size_t)l * G4, bhh + (size_t)l * G4, xproj);

    reset_bar<<<1, 1, 0, stream>>>(bar);

    lstm_scan<<<HH / NSLICE, 128, 0, stream>>>(
        whhb, xproj, h0, h1,
        last ? out : nullptr,
        last ? nullptr : actout,
        bar);
  }
}
